// NeuralNetSinkhornPRModel_56358560858147
// MI455X (gfx1250) — matrix operand order repaired and checked by reading, not compiled, not run on hardware
//
#include <hip/hip_runtime.h>
#include <hip/hip_bf16.h>
#include <math.h>

// ---------------------------------------------------------------------------
// Types for CDNA5 WMMA (gfx1250, wave32)
// ---------------------------------------------------------------------------
typedef __attribute__((ext_vector_type(16))) __bf16 v16bf;
typedef __attribute__((ext_vector_type(8)))  __bf16 v8bf;
typedef __attribute__((ext_vector_type(8)))  float  v8f;

// Problem constants (from reference)
#define ND   4096   // rows of d
#define NS   2048   // rows of s1/s2
#define DIN  768    // input feature dim
#define DH   128    // hidden dim (W rows)
#define NSTEPS 20
#define EPS_TGT 0.0025f   // blur^p = 0.05^2
#define QRATIO  0.25f     // scaling^p = 0.5^2

// ---------------------------------------------------------------------------
// wave32 reduction helpers (no LDS, no barrier inside a wave)
// ---------------------------------------------------------------------------
__device__ __forceinline__ float wave_max(float v) {
#pragma unroll
  for (int off = 16; off > 0; off >>= 1) v = fmaxf(v, __shfl_xor(v, off, 32));
  return v;
}
__device__ __forceinline__ float wave_sum(float v) {
#pragma unroll
  for (int off = 16; off > 0; off >>= 1) v += __shfl_xor(v, off, 32);
  return v;
}

// ---------------------------------------------------------------------------
// Split f32 -> bf16 hi + bf16 lo (error-compensated representation)
// ---------------------------------------------------------------------------
__global__ void split_bf16_kernel(const float* __restrict__ x, size_t n,
                                  __bf16* __restrict__ hi, __bf16* __restrict__ lo) {
  size_t i = (size_t)blockIdx.x * 256 + threadIdx.x;
  if (i < n) {
    float v = x[i];
    __bf16 h = (__bf16)v;
    float  hf = (float)h;
    hi[i] = h;
    lo[i] = (__bf16)(v - hf);
  }
}

// ---------------------------------------------------------------------------
// WMMA GEMM:  C[M,N] = A[M,K] @ B[N,K]^T   (A,B given as bf16 hi/lo splits)
// 3-term compensation: hi*hi + hi*lo + lo*hi  (f32 accumulate)
// Block = 256 threads = 8 waves; wave tile 16x16; block tile 32(M) x 64(N).
// Requires M%32==0, N%64==0, K%32==0 (true for all call sites here).
// Optional epilogue: +bias, relu.
// ---------------------------------------------------------------------------
__global__ __launch_bounds__(256)
void wmma_gemm_nt_kernel(const __bf16* __restrict__ Ahi, const __bf16* __restrict__ Alo,
                         const __bf16* __restrict__ Bhi, const __bf16* __restrict__ Blo,
                         float* __restrict__ C, int M, int N, int K,
                         const float* __restrict__ bias, int relu) {
  const int lane  = threadIdx.x & 31;
  const int wave  = threadIdx.x >> 5;
  const int wm    = wave >> 2;          // 0..1
  const int wn    = wave & 3;           // 0..3
  const int tile_m = blockIdx.y * 32 + wm * 16;
  const int tile_n = blockIdx.x * 64 + wn * 16;
  if (tile_m >= M || tile_n >= N) return;

  const int lhalf = lane >> 4;          // 0/1
  const int l15   = lane & 15;
  const int ag    = 8 * lhalf;          // A K-group base within 32-chunk

  const __bf16* arow_hi = Ahi + (size_t)(tile_m + l15) * K;
  const __bf16* arow_lo = Alo + (size_t)(tile_m + l15) * K;
  const __bf16* brow_hi = Bhi + (size_t)(tile_n + l15) * K;
  const __bf16* brow_lo = Blo + (size_t)(tile_n + l15) * K;

  v8f acc0 = {}; v8f acc1 = {}; v8f acc2 = {};

  for (int k0 = 0; k0 < K; k0 += 32) {
    // A fragment (16x32 bf16): lane holds row M=l15, K = {ag..ag+7} U {16+ag..16+ag+7}
    v8bf a0h = *(const v8bf*)(arow_hi + k0 + ag);
    v8bf a1h = *(const v8bf*)(arow_hi + k0 + ag + 16);
    v8bf a0l = *(const v8bf*)(arow_lo + k0 + ag);
    v8bf a1l = *(const v8bf*)(arow_lo + k0 + ag + 16);
    // B fragment (32x16 bf16): lane holds col N=l15, K = {ag..ag+7} U {16+ag..16+ag+7}
    v8bf b0h = *(const v8bf*)(brow_hi + k0 + ag);
    v8bf b1h = *(const v8bf*)(brow_hi + k0 + ag + 16);
    v8bf b0l = *(const v8bf*)(brow_lo + k0 + ag);
    v8bf b1l = *(const v8bf*)(brow_lo + k0 + ag + 16);

    if (k0 + 32 < K) {   // WGP-scope prefetch (fills near caches) for next K slab
      __builtin_prefetch(arow_hi + k0 + 32 + ag, 0, 3);
      __builtin_prefetch(brow_hi + k0 + 32 + 16 * lhalf, 0, 3);
    }

    v16bf ah, al, bh, bl;
#pragma unroll
    for (int e = 0; e < 8; ++e) {
      ah[e] = a0h[e]; ah[8 + e] = a1h[e];
      al[e] = a0l[e]; al[8 + e] = a1l[e];
      bh[e] = b0h[e]; bh[8 + e] = b1h[e];
      bl[e] = b0l[e]; bl[8 + e] = b1l[e];
    }
    acc0 = __builtin_amdgcn_wmma_f32_16x16x32_bf16(false, ah, false, bh, (short)0, acc0, false, false);
    acc1 = __builtin_amdgcn_wmma_f32_16x16x32_bf16(false, ah, false, bl, (short)0, acc1, false, false);
    acc2 = __builtin_amdgcn_wmma_f32_16x16x32_bf16(false, al, false, bh, (short)0, acc2, false, false);
  }

  // C/D layout: VGPR v, lanes 0-15 -> M=v, lanes 16-31 -> M=v+8 ; N = lane&15
  const int cm = tile_m + lhalf * 8;
  const int cn = tile_n + l15;
  const float bval = bias ? bias[cn] : 0.0f;
#pragma unroll
  for (int v = 0; v < 8; ++v) {
    float val = acc0[v] + acc1[v] + acc2[v] + bval;
    if (relu) val = fmaxf(val, 0.0f);
    C[(size_t)(cm + v) * N + cn] = val;
  }
}

// ---------------------------------------------------------------------------
// Post-transform: bf16 split of the transformed features + row squared norms
// One block (128 threads = 4 waves) per row, D = 128.
// ---------------------------------------------------------------------------
__global__ __launch_bounds__(128)
void post_transform_kernel(const float* __restrict__ T,
                           __bf16* __restrict__ hi, __bf16* __restrict__ lo,
                           float* __restrict__ x2) {
  __shared__ float wred[4];
  const int i = blockIdx.x, t = threadIdx.x;
  const int lane = t & 31, wave = t >> 5;
  float v = T[(size_t)i * DH + t];
  __bf16 h = (__bf16)v; float hf = (float)h;
  hi[(size_t)i * DH + t] = h;
  lo[(size_t)i * DH + t] = (__bf16)(v - hf);
  float s = wave_sum(v * v);
  if (lane == 0) wred[wave] = s;
  __syncthreads();
  if (t == 0) x2[i] = wred[0] + wred[1] + wred[2] + wred[3];
}

// ---------------------------------------------------------------------------
// Per-dim squared range over union of X (Mx rows) and Y (My rows); blockIdx = dim
// ---------------------------------------------------------------------------
__global__ __launch_bounds__(256)
void range_kernel(const float* __restrict__ X, int Mx,
                  const float* __restrict__ Y, int My,
                  float* __restrict__ rng) {
  __shared__ float wmn[8], wmx[8];
  const int dim = blockIdx.x, t = threadIdx.x;
  const int lane = t & 31, wave = t >> 5;
  float mn = INFINITY, mx = -INFINITY;
  for (int r = t; r < Mx; r += 256) { float v = X[(size_t)r * DH + dim]; mn = fminf(mn, v); mx = fmaxf(mx, v); }
  for (int r = t; r < My; r += 256) { float v = Y[(size_t)r * DH + dim]; mn = fminf(mn, v); mx = fmaxf(mx, v); }
  mx = wave_max(mx);
  mn = -wave_max(-mn);
  if (lane == 0) { wmn[wave] = mn; wmx[wave] = mx; }
  __syncthreads();
  if (t == 0) {
#pragma unroll
    for (int w = 1; w < 8; ++w) { wmn[0] = fminf(wmn[0], wmn[w]); wmx[0] = fmaxf(wmx[0], wmx[w]); }
    float r = wmx[0] - wmn[0];
    rng[dim] = r * r;
  }
}

__global__ __launch_bounds__(128)
void eps0_kernel(const float* __restrict__ rng, float* __restrict__ eps0) {
  __shared__ float wred[4];
  const int t = threadIdx.x, lane = t & 31, wave = t >> 5;
  float s = wave_sum(rng[t]);
  if (lane == 0) wred[wave] = s;
  __syncthreads();
  if (t == 0) eps0[0] = fmaxf(wred[0] + wred[1] + wred[2] + wred[3], EPS_TGT); // diam^2 clamp
}

// ---------------------------------------------------------------------------
// Row softmin:  out[i] = -eps * LSE_j( logw + pot[j]/eps - C(i,j)/eps )
//   C(i,j) = max(0.5*(x2[i]+y2[j]) - G[i,j], 0)
// eps = max(eps0*qk, EPS_TGT) read from device scalar (graph-capture safe).
// One block (256 threads = 8 waves) per row; CPT = cols/256 values in regs:
// one global read + one v_exp_f32 per element; wave32 shuffle reductions.
// ---------------------------------------------------------------------------
template <int CPT>
__global__ __launch_bounds__(256)
void softmin_rows_kernel(const float* __restrict__ G, int ldg,
                         const float* __restrict__ x2, const float* __restrict__ y2,
                         const float* __restrict__ pot, float logw,
                         const float* __restrict__ eps0_ptr, float qk,
                         float* __restrict__ outv) {
  __shared__ float wmaxs[8], wsums[8];
  const int i = blockIdx.x, tid = threadIdx.x;
  const int lane = tid & 31, wave = tid >> 5;
  const float eps = fmaxf(eps0_ptr[0] * qk, EPS_TGT);
  const float inv_eps = 1.0f / eps;
  const float xi = 0.5f * x2[i];
  const float* grow = G + (size_t)i * ldg;

  float vloc[CPT];
  float m = -INFINITY;
#pragma unroll
  for (int u = 0; u < CPT; ++u) {
    const int j = tid + u * 256;
    float c = fmaxf(xi + 0.5f * y2[j] - grow[j], 0.0f);
    float h = logw;
    if (pot) h += pot[j] * inv_eps;
    float v = h - c * inv_eps;
    vloc[u] = v;
    m = fmaxf(m, v);
  }
  // block max: intra-wave butterfly + 8-entry cross-wave combine
  m = wave_max(m);
  if (lane == 0) wmaxs[wave] = m;
  __syncthreads();
  m = wmaxs[0];
#pragma unroll
  for (int w = 1; w < 8; ++w) m = fmaxf(m, wmaxs[w]);

  float ssum = 0.0f;
#pragma unroll
  for (int u = 0; u < CPT; ++u) ssum += __expf(vloc[u] - m);
  ssum = wave_sum(ssum);
  if (lane == 0) wsums[wave] = ssum;
  __syncthreads();
  if (tid == 0) {
    float S = wsums[0];
#pragma unroll
    for (int w = 1; w < 8; ++w) S += wsums[w];
    outv[i] = -eps * (m + __logf(S));
  }
}

// f = 0.5*(f + t)
__global__ void avg_kernel(float* __restrict__ f, const float* __restrict__ t, int n) {
  int i = blockIdx.x * 256 + threadIdx.x;
  if (i < n) f[i] = 0.5f * (f[i] + t[i]);
}

// dist = mean(ft - fa) + mean(gt - gb)
__global__ __launch_bounds__(256)
void dist_kernel(const float* __restrict__ ft, const float* __restrict__ fa, int n1,
                 const float* __restrict__ gt, const float* __restrict__ gb, int n2,
                 float* __restrict__ dist_out) {
  __shared__ float w1[8], w2[8];
  const int t = threadIdx.x, lane = t & 31, wave = t >> 5;
  float s1 = 0.f, s2 = 0.f;
  for (int i = t; i < n1; i += 256) s1 += ft[i] - fa[i];
  for (int i = t; i < n2; i += 256) s2 += gt[i] - gb[i];
  s1 = wave_sum(s1); s2 = wave_sum(s2);
  if (lane == 0) { w1[wave] = s1; w2[wave] = s2; }
  __syncthreads();
  if (t == 0) {
    float S1 = 0.f, S2 = 0.f;
#pragma unroll
    for (int w = 0; w < 8; ++w) { S1 += w1[w]; S2 += w2[w]; }
    dist_out[0] = S1 / (float)n1 + S2 / (float)n2;
  }
}

__global__ void sigmoid_kernel(const float* __restrict__ dist, float* __restrict__ out) {
  if (threadIdx.x == 0) out[0] = 1.0f / (1.0f + __expf(-(dist[1] - dist[0])));
}

// ---------------------------------------------------------------------------
// Host orchestration
// ---------------------------------------------------------------------------
static inline void launch_softmin(hipStream_t st, const float* G, int rows, int cols,
                                  const float* x2, const float* y2, const float* pot,
                                  float logw, const float* eps0, float qk, float* out) {
  if (cols == 2048)
    softmin_rows_kernel<8><<<rows, 256, 0, st>>>(G, cols, x2, y2, pot, logw, eps0, qk, out);
  else
    softmin_rows_kernel<16><<<rows, 256, 0, st>>>(G, cols, x2, y2, pot, logw, eps0, qk, out);
}

extern "C" void kernel_launch(void* const* d_in, const int* in_sizes, int n_in,
                              void* d_out, int out_size, void* d_ws, size_t ws_size,
                              hipStream_t stream) {
  (void)in_sizes; (void)n_in; (void)out_size; (void)ws_size;
  const float* d  = (const float*)d_in[0];   // 4096 x 768
  const float* s1 = (const float*)d_in[1];   // 2048 x 768
  const float* s2 = (const float*)d_in[2];   // 2048 x 768
  const float* W  = (const float*)d_in[3];   // 128  x 768
  const float* bb = (const float*)d_in[4];   // 128
  float* out = (float*)d_out;

  // ---- workspace bump allocator (256B aligned) ----
  char* p = (char*)d_ws;
  auto alloc = [&](size_t bytes) -> char* {
    char* r = p; p += (bytes + 255) & ~(size_t)255; return r;
  };
  __bf16* dhi = (__bf16*)alloc((size_t)ND * DIN * 2);
  __bf16* dlo = (__bf16*)alloc((size_t)ND * DIN * 2);
  __bf16* whi = (__bf16*)alloc((size_t)DH * DIN * 2);
  __bf16* wlo = (__bf16*)alloc((size_t)DH * DIN * 2);
  __bf16* shi = (__bf16*)alloc((size_t)NS * DIN * 2);   // reused for s1 then s2
  __bf16* slo = (__bf16*)alloc((size_t)NS * DIN * 2);
  float* td = (float*)alloc((size_t)ND * DH * 4);
  float* t1 = (float*)alloc((size_t)NS * DH * 4);
  float* t2 = (float*)alloc((size_t)NS * DH * 4);
  __bf16* tdhi = (__bf16*)alloc((size_t)ND * DH * 2);
  __bf16* tdlo = (__bf16*)alloc((size_t)ND * DH * 2);
  __bf16* t1hi = (__bf16*)alloc((size_t)NS * DH * 2);
  __bf16* t1lo = (__bf16*)alloc((size_t)NS * DH * 2);
  __bf16* t2hi = (__bf16*)alloc((size_t)NS * DH * 2);
  __bf16* t2lo = (__bf16*)alloc((size_t)NS * DH * 2);
  float* x2d = (float*)alloc((size_t)ND * 4);
  float* y21 = (float*)alloc((size_t)NS * 4);
  float* y22 = (float*)alloc((size_t)NS * 4);
  float* Gxx = (float*)alloc((size_t)ND * ND * 4);
  float* Gxy = (float*)alloc((size_t)ND * NS * 4);
  float* Gyx = (float*)alloc((size_t)NS * ND * 4);
  float* Gyy = (float*)alloc((size_t)NS * NS * 4);
  float* f_ab = (float*)alloc((size_t)ND * 4);
  float* g_ab = (float*)alloc((size_t)NS * 4);
  float* f_aa = (float*)alloc((size_t)ND * 4);
  float* g_bb = (float*)alloc((size_t)NS * 4);
  float* tmpf = (float*)alloc((size_t)ND * 4);
  float* tmpg = (float*)alloc((size_t)NS * 4);
  float* ftv  = (float*)alloc((size_t)ND * 4);
  float* gtv  = (float*)alloc((size_t)NS * 4);
  float* fav  = (float*)alloc((size_t)ND * 4);
  float* gbv  = (float*)alloc((size_t)NS * 4);
  float* rng  = (float*)alloc((size_t)DH * 4);
  float* eps0 = (float*)alloc(256);
  float* dist = (float*)alloc(256);

  const float LOGA = -logf((float)ND);   // log_a (uniform)
  const float LOGB = -logf((float)NS);   // log_b (uniform)

  // ---- transform: t = relu(x @ W^T + b) (WMMA, bf16x3) ----
  split_bf16_kernel<<<(ND * DIN + 255) / 256, 256, 0, stream>>>(d, (size_t)ND * DIN, dhi, dlo);
  split_bf16_kernel<<<(DH * DIN + 255) / 256, 256, 0, stream>>>(W, (size_t)DH * DIN, whi, wlo);
  wmma_gemm_nt_kernel<<<dim3(DH / 64, ND / 32), 256, 0, stream>>>(dhi, dlo, whi, wlo, td, ND, DH, DIN, bb, 1);
  post_transform_kernel<<<ND, 128, 0, stream>>>(td, tdhi, tdlo, x2d);

  split_bf16_kernel<<<(NS * DIN + 255) / 256, 256, 0, stream>>>(s1, (size_t)NS * DIN, shi, slo);
  wmma_gemm_nt_kernel<<<dim3(DH / 64, NS / 32), 256, 0, stream>>>(shi, slo, whi, wlo, t1, NS, DH, DIN, bb, 1);
  post_transform_kernel<<<NS, 128, 0, stream>>>(t1, t1hi, t1lo, y21);

  split_bf16_kernel<<<(NS * DIN + 255) / 256, 256, 0, stream>>>(s2, (size_t)NS * DIN, shi, slo);
  wmma_gemm_nt_kernel<<<dim3(DH / 64, NS / 32), 256, 0, stream>>>(shi, slo, whi, wlo, t2, NS, DH, DIN, bb, 1);
  post_transform_kernel<<<NS, 128, 0, stream>>>(t2, t2hi, t2lo, y22);

  // ---- shared Gram: G_xx = td @ td^T ----
  wmma_gemm_nt_kernel<<<dim3(ND / 64, ND / 32), 256, 0, stream>>>(tdhi, tdlo, tdhi, tdlo, Gxx, ND, ND, DH, nullptr, 0);

  for (int v = 0; v < 2; ++v) {
    const float* ty = v ? t2 : t1;
    const __bf16* tyhi = v ? t2hi : t1hi;
    const __bf16* tylo = v ? t2lo : t1lo;
    const float* y2 = v ? y22 : y21;

    // Gram matrices for this divergence
    wmma_gemm_nt_kernel<<<dim3(NS / 64, ND / 32), 256, 0, stream>>>(tdhi, tdlo, tyhi, tylo, Gxy, ND, NS, DH, nullptr, 0);
    wmma_gemm_nt_kernel<<<dim3(ND / 64, NS / 32), 256, 0, stream>>>(tyhi, tylo, tdhi, tdlo, Gyx, NS, ND, DH, nullptr, 0);
    wmma_gemm_nt_kernel<<<dim3(NS / 64, NS / 32), 256, 0, stream>>>(tyhi, tylo, tyhi, tylo, Gyy, NS, NS, DH, nullptr, 0);

    // eps0 = max(diam^2, eps_tgt), computed on device (graph-safe)
    range_kernel<<<DH, 256, 0, stream>>>(td, ND, ty, NS, rng);
    eps0_kernel<<<1, 128, 0, stream>>>(rng, eps0);

    // init potentials at eps0
    launch_softmin(stream, Gxy, ND, NS, x2d, y2, nullptr, LOGB, eps0, 1.0f, f_ab);
    launch_softmin(stream, Gyx, NS, ND, y2, x2d, nullptr, LOGA, eps0, 1.0f, g_ab);
    launch_softmin(stream, Gxx, ND, ND, x2d, x2d, nullptr, LOGA, eps0, 1.0f, f_aa);
    launch_softmin(stream, Gyy, NS, NS, y2, y2, nullptr, LOGB, eps0, 1.0f, g_bb);

    // eps-annealed symmetric Sinkhorn iterations
    float qk = 1.0f;
    for (int k = 0; k < NSTEPS; ++k) {
      launch_softmin(stream, Gxy, ND, NS, x2d, y2, g_ab, LOGB, eps0, qk, tmpf);
      launch_softmin(stream, Gyx, NS, ND, y2, x2d, f_ab, LOGA, eps0, qk, tmpg);
      avg_kernel<<<(ND + 255) / 256, 256, 0, stream>>>(f_ab, tmpf, ND);
      avg_kernel<<<(NS + 255) / 256, 256, 0, stream>>>(g_ab, tmpg, NS);
      launch_softmin(stream, Gxx, ND, ND, x2d, x2d, f_aa, LOGA, eps0, qk, tmpf);
      avg_kernel<<<(ND + 255) / 256, 256, 0, stream>>>(f_aa, tmpf, ND);
      launch_softmin(stream, Gyy, NS, NS, y2, y2, g_bb, LOGB, eps0, qk, tmpg);
      avg_kernel<<<(NS + 255) / 256, 256, 0, stream>>>(g_bb, tmpg, NS);
      qk *= QRATIO;
    }

    // final extrapolation at eps = blur^p (qk=0 -> eps clamps to EPS_TGT)
    launch_softmin(stream, Gxy, ND, NS, x2d, y2, g_ab, LOGB, eps0, 0.0f, ftv);
    launch_softmin(stream, Gyx, NS, ND, y2, x2d, f_ab, LOGA, eps0, 0.0f, gtv);
    launch_softmin(stream, Gxx, ND, ND, x2d, x2d, f_aa, LOGA, eps0, 0.0f, fav);
    launch_softmin(stream, Gyy, NS, NS, y2, y2, g_bb, LOGB, eps0, 0.0f, gbv);
    dist_kernel<<<1, 256, 0, stream>>>(ftv, fav, ND, gtv, gbv, NS, dist + v);
  }

  sigmoid_kernel<<<1, 32, 0, stream>>>(dist, out);
}
